// MultiHeadAttention_52767968198874
// MI455X (gfx1250) — matrix operand order repaired and checked by reading, not compiled, not run on hardware
//
#include <hip/hip_runtime.h>

// ---------------------------------------------------------------------------
// MHA forward for MI455X (gfx1250, wave32, WMMA bf16).
// Scores computed TRANSPOSED (S^T = K Q^T): softmax reductions are in-lane
// (+1 xor-16 shuffle); P relayout and output stores are packed 16-byte ops.
// Attention uses a 64-key step (16 WMMAs per softmax pass), raw v_exp_f32,
// double-buffered LDS staging; GEMMs compute C^T tiles with packed epilogues.
// ---------------------------------------------------------------------------

typedef __attribute__((ext_vector_type(16))) __bf16 v16bf;
typedef __attribute__((ext_vector_type(8)))  __bf16 v8bf;
typedef __attribute__((ext_vector_type(8)))  float  v8f;

#define DMODEL 1024
#define SEQ    2048
#define NBATCH 2
#define NHEADS 16
#define HDIM   64
#define MTOT   (NBATCH*SEQ)   // 4096

__device__ __forceinline__ v16bf frag_join(const __bf16* p0, const __bf16* p1) {
  v8bf a = *(const v8bf*)p0;
  v8bf b = *(const v8bf*)p1;
  return __builtin_shufflevector(a, b, 0,1,2,3,4,5,6,7,8,9,10,11,12,13,14,15);
}
// A-fragment (16x32): lane row = lane%16, K chunks {8h, 16+8h}
__device__ __forceinline__ v16bf afrag(const __bf16* row, int hf) {
  return frag_join(row + 8 * hf, row + 16 + 8 * hf);
}
// B-fragment (32x16) from B^T-major row: lane col = lane%16, K chunks {8h, 16+8h}
__device__ __forceinline__ v16bf bfrag(const __bf16* row, int hf) {
  return frag_join(row + 8 * hf, row + 16 + 8 * hf);
}
__device__ __forceinline__ v8f wmma_bf16(v16bf a, v16bf b, v8f c) {
  return __builtin_amdgcn_wmma_f32_16x16x32_bf16(false, a, false, b, (short)0, c,
                                                 false, false);
}

// -------------------------- fp32 -> bf16 convert ---------------------------
__global__ __launch_bounds__(256)
void mha_cvt_bf16(const float* __restrict__ in, __bf16* __restrict__ out) {
  int i = (blockIdx.x * 256 + threadIdx.x) * 8;
  float4 a = *(const float4*)&in[i];
  float4 b = *(const float4*)&in[i + 4];
  v8bf r;
  r[0] = (__bf16)a.x; r[1] = (__bf16)a.y; r[2] = (__bf16)a.z; r[3] = (__bf16)a.w;
  r[4] = (__bf16)b.x; r[5] = (__bf16)b.y; r[6] = (__bf16)b.z; r[7] = (__bf16)b.w;
  *(v8bf*)&out[i] = r;
}

// ------------------- transpose + convert: Wt[n,k] = W[k,n] -----------------
__global__ __launch_bounds__(256)
void mha_transpose_cvt(const float* __restrict__ W, __bf16* __restrict__ Wt) {
  __shared__ float tile[32][33];
  int n0 = blockIdx.x * 32;
  int k0 = blockIdx.y * 32;
  int tx = threadIdx.x;
  #pragma unroll
  for (int i = threadIdx.y; i < 32; i += 8)
    tile[i][tx] = W[(k0 + i) * DMODEL + n0 + tx];
  __syncthreads();
  #pragma unroll
  for (int i = threadIdx.y; i < 32; i += 8)
    Wt[(n0 + i) * DMODEL + k0 + tx] = (__bf16)tile[tx][i];
}

// ------------------------------- WMMA GEMM ---------------------------------
// out(M,1024) = A(M,1024) * Bt(1024,1024)^T + bias.  Computed as C^T tiles
// (n in registers, m in lanes) so epilogue stores are packed.  Block 128x128,
// 8 waves = (wm 4) x (wn 2), wave tile 32(m) x 64(n), K-step 32, double-buffered.
enum { OUT_BF16 = 0, OUT_VT = 1, OUT_F32 = 2 };

template <int MODE>
__global__ __launch_bounds__(256)
void mha_gemm_bf16(const __bf16* __restrict__ A, const __bf16* __restrict__ Bt,
                   const float* __restrict__ bias, void* __restrict__ out) {
  constexpr int LD = 40;
  __shared__ __bf16 As[2][128 * LD];
  __shared__ __bf16 Bs[2][128 * LD];

  const int tid  = threadIdx.x;
  const int lane = tid & 31;
  const int wave = tid >> 5;
  const int l16  = lane & 15;
  const int hf   = lane >> 4;
  const int M0   = blockIdx.y * 128;
  const int N0   = blockIdx.x * 128;
  const int wm   = wave & 3;
  const int wn   = wave >> 2;

  const int r0 = tid >> 2;                // 0..63
  const int ko = (tid & 3) * 8;

  v8bf ra0, ra1, rb0, rb1;
  auto gload = [&](int k0) {
    ra0 = *(const v8bf*)&A [(M0 + r0     ) * DMODEL + k0 + ko];
    ra1 = *(const v8bf*)&A [(M0 + r0 + 64) * DMODEL + k0 + ko];
    rb0 = *(const v8bf*)&Bt[(N0 + r0     ) * DMODEL + k0 + ko];
    rb1 = *(const v8bf*)&Bt[(N0 + r0 + 64) * DMODEL + k0 + ko];
  };
  auto sstore = [&](int buf) {
    *(v8bf*)&As[buf][(r0     ) * LD + ko] = ra0;
    *(v8bf*)&As[buf][(r0 + 64) * LD + ko] = ra1;
    *(v8bf*)&Bs[buf][(r0     ) * LD + ko] = rb0;
    *(v8bf*)&Bs[buf][(r0 + 64) * LD + ko] = rb1;
  };

  v8f acc[4][2] = {};
  gload(0); sstore(0);
  int cur = 0;

  for (int k0 = 0; k0 < DMODEL; k0 += 32) {
    __syncthreads();
    const bool more = (k0 + 32) < DMODEL;
    if (more) gload(k0 + 32);

    v16bf af[4], bfm[2];
    #pragma unroll
    for (int i = 0; i < 4; ++i)
      af[i] = afrag(&Bs[cur][(wn * 64 + i * 16 + l16) * LD], hf);
    #pragma unroll
    for (int j = 0; j < 2; ++j)
      bfm[j] = bfrag(&As[cur][(wm * 32 + j * 16 + l16) * LD], hf);
    #pragma unroll
    for (int i = 0; i < 4; ++i)
      #pragma unroll
      for (int j = 0; j < 2; ++j)
        acc[i][j] = wmma_bf16(af[i], bfm[j], acc[i][j]);

    if (more) sstore(cur ^ 1);
    cur ^= 1;
  }

  #pragma unroll
  for (int i = 0; i < 4; ++i) {
    const int nb = N0 + wn * 64 + i * 16 + 8 * hf;
    float4 bv0 = *(const float4*)&bias[nb];
    float4 bv1 = *(const float4*)&bias[nb + 4];
    const float bval[8] = {bv0.x, bv0.y, bv0.z, bv0.w, bv1.x, bv1.y, bv1.z, bv1.w};
    #pragma unroll
    for (int j = 0; j < 2; ++j) {
      const int m = M0 + wm * 32 + j * 16 + l16;
      if (MODE == OUT_F32) {
        float4 o0, o1;
        o0.x = acc[i][j][0] + bval[0]; o0.y = acc[i][j][1] + bval[1];
        o0.z = acc[i][j][2] + bval[2]; o0.w = acc[i][j][3] + bval[3];
        o1.x = acc[i][j][4] + bval[4]; o1.y = acc[i][j][5] + bval[5];
        o1.z = acc[i][j][6] + bval[6]; o1.w = acc[i][j][7] + bval[7];
        *(float4*)&((float*)out)[m * DMODEL + nb]     = o0;
        *(float4*)&((float*)out)[m * DMODEL + nb + 4] = o1;
      } else if (MODE == OUT_BF16) {
        v8bf r;
        #pragma unroll
        for (int v = 0; v < 8; ++v) r[v] = (__bf16)(acc[i][j][v] + bval[v]);
        *(v8bf*)&((__bf16*)out)[m * DMODEL + nb] = r;
      } else {                           // OUT_VT: V stored (b,h,dh,s)
        const int b = m >> 11, s = m & (SEQ - 1);
        #pragma unroll
        for (int v = 0; v < 8; ++v) {
          int n = nb + v;
          int hh = n >> 6, dh = n & (HDIM - 1);
          ((__bf16*)out)[((b * NHEADS + hh) * HDIM + dh) * SEQ + s] =
              (__bf16)(acc[i][j][v] + bval[v]);
        }
      }
    }
  }
}

// ---------------------------- flash attention ------------------------------
// Grid (SEQ/128, NBATCH*NHEADS); 8 waves each own 16 queries; 64-key steps.
// Per step: S^T = K Q^T (8 WMMA), per-lane online softmax (raw v_exp_f32),
// P via packed LDS bounce, O^T += V^T P (8 WMMA).
__global__ __launch_bounds__(256)
void mha_attention(const __bf16* __restrict__ Q,   // (4096,1024)
                   const __bf16* __restrict__ Kg,  // (4096,1024)
                   const __bf16* __restrict__ Vt,  // (bh,dh,s) = (32,64,2048)
                   __bf16* __restrict__ O) {       // (4096,1024)
  constexpr int KLD = 72, VLD = 72, PLD = 72;      // 64 + 8 pad
  __shared__ __bf16 Kt[2][64 * KLD];               // [key][dh]
  __shared__ __bf16 Vs[2][64 * VLD];               // [dh][key]
  __shared__ __bf16 Ps[8][16 * PLD];               // per-wave [query][key]

  const int tid  = threadIdx.x;
  const int lane = tid & 31;
  const int wave = tid >> 5;
  const int l16  = lane & 15;
  const int hf   = lane >> 4;
  const int bh = blockIdx.y;
  const int b  = bh >> 4, h = bh & 15;
  const int q0 = blockIdx.x * 128;
  const int qrow = b * SEQ + q0 + wave * 16 + l16;

  const float CEXP = 0.125f * 1.44269504088896340736f;  // scale * log2(e)

  // Q as B-fragments (B = Q^T): lane = query, k = dh; held for whole loop.
  v16bf qf[2];
  #pragma unroll
  for (int ks = 0; ks < 2; ++ks)
    qf[ks] = bfrag(&Q[qrow * DMODEL + h * HDIM + ks * 32], hf);

  v8f oacc[4] = {};                        // O^T tiles: dh in regs, q in lanes
  float mrun = -3.0e38f, lrun = 0.0f;      // per-lane (per-query) state

  // staging: K 64x64 and V^T 64x64, each thread 2 chunks per tile
  const int sr = tid >> 3;                 // 0..31 (and +32)
  const int sc = (tid & 7) * 8;            // 0..56
  const __bf16* kgp = &Kg[(b * SEQ + sr) * DMODEL + h * HDIM + sc];
  const __bf16* vgp = &Vt[(bh * HDIM + sr) * SEQ + sc];

  v8bf kr0, kr1, vr0, vr1;
  auto gload = [&](int k0) {
    kr0 = *(const v8bf*)&kgp[(size_t)(k0     ) * DMODEL];
    kr1 = *(const v8bf*)&kgp[(size_t)(k0 + 32) * DMODEL];
    vr0 = *(const v8bf*)&vgp[k0];                       // dh row sr
    vr1 = *(const v8bf*)&vgp[(size_t)32 * SEQ + k0];    // dh row sr+32
  };
  auto sstore = [&](int buf) {
    *(v8bf*)&Kt[buf][(sr     ) * KLD + sc] = kr0;
    *(v8bf*)&Kt[buf][(sr + 32) * KLD + sc] = kr1;
    *(v8bf*)&Vs[buf][(sr     ) * VLD + sc] = vr0;
    *(v8bf*)&Vs[buf][(sr + 32) * VLD + sc] = vr1;
  };

  gload(0); sstore(0);
  int cur = 0;

  for (int k0 = 0; k0 < SEQ; k0 += 64) {
    __syncthreads();
    const bool more = (k0 + 64) < SEQ;
    if (more) gload(k0 + 64);

    // S^T = K Q^T : keys (4 tiles) in regs, queries in lanes
    v8f sac[4] = {};
    #pragma unroll
    for (int i = 0; i < 4; ++i)
      #pragma unroll
      for (int ks = 0; ks < 2; ++ks) {
        v16bf kf = afrag(&Kt[cur][(i * 16 + l16) * KLD + ks * 32], hf);
        sac[i] = wmma_bf16(kf, qf[ks], sac[i]);
      }

    // preload all V^T fragments (LDS pipe runs ahead of softmax VALU)
    v16bf vf[4][2];
    #pragma unroll
    for (int j = 0; j < 4; ++j)
      #pragma unroll
      for (int kk = 0; kk < 2; ++kk)
        vf[j][kk] = afrag(&Vs[cur][(j * 16 + l16) * VLD + kk * 32], hf);

    // per-query softmax: in-lane tree + one xor-16 combine; raw v_exp_f32
    float tm = fmaxf(fmaxf(sac[0][0], sac[1][0]), fmaxf(sac[2][0], sac[3][0]));
    #pragma unroll
    for (int v = 1; v < 8; ++v)
      tm = fmaxf(tm, fmaxf(fmaxf(sac[0][v], sac[1][v]),
                           fmaxf(sac[2][v], sac[3][v])));
    tm = fmaxf(tm, __shfl_xor(tm, 16, 32));
    const float mnew  = fmaxf(mrun, tm);
    const float alpha = __builtin_amdgcn_exp2f((mrun - mnew) * CEXP);
    mrun = mnew;
    const float msub = mnew * CEXP;
    v8bf pb[4];
    float rs = 0.0f;
    #pragma unroll
    for (int i = 0; i < 4; ++i)
      #pragma unroll
      for (int v = 0; v < 8; ++v) {
        float p = __builtin_amdgcn_exp2f(sac[i][v] * CEXP - msub);
        rs += p;
        pb[i][v] = (__bf16)p;
      }
    rs += __shfl_xor(rs, 16, 32);
    lrun = lrun * alpha + rs;
    #pragma unroll
    for (int j = 0; j < 4; ++j)
      #pragma unroll
      for (int v = 0; v < 8; ++v) oacc[j][v] *= alpha;

    // P^T (C-layout) -> [query][key] rows via packed 16B LDS stores
    __bf16* ps = &Ps[wave][0];
    #pragma unroll
    for (int i = 0; i < 4; ++i)
      *(v8bf*)&ps[l16 * PLD + i * 16 + 8 * hf] = pb[i];
    v16bf pf[2];
    #pragma unroll
    for (int kk = 0; kk < 2; ++kk)
      pf[kk] = bfrag(&ps[l16 * PLD + kk * 32], hf);

    // O^T += V^T P : 4 dh tiles x 2 key-chunks
    #pragma unroll
    for (int j = 0; j < 4; ++j)
      #pragma unroll
      for (int kk = 0; kk < 2; ++kk)
        oacc[j] = wmma_bf16(vf[j][kk], pf[kk], oacc[j]);

    if (more) sstore(cur ^ 1);
    cur ^= 1;
  }

  // normalize + packed 16B stores: per lane, 8 consecutive dh per tile
  const float inv = __builtin_amdgcn_rcpf(lrun);
  #pragma unroll
  for (int j = 0; j < 4; ++j) {
    v8bf r;
    #pragma unroll
    for (int v = 0; v < 8; ++v) r[v] = (__bf16)(oacc[j][v] * inv);
    *(v8bf*)&O[qrow * DMODEL + h * HDIM + j * 16 + 8 * hf] = r;
  }
}

// ------------------------------- launcher ----------------------------------
extern "C" void kernel_launch(void* const* d_in, const int* in_sizes, int n_in,
                              void* d_out, int out_size, void* d_ws, size_t ws_size,
                              hipStream_t stream) {
  const float* x  = (const float*)d_in[0];
  const float* Wq = (const float*)d_in[1];
  const float* bq = (const float*)d_in[2];
  const float* Wk = (const float*)d_in[3];
  const float* bk = (const float*)d_in[4];
  const float* Wv = (const float*)d_in[5];
  const float* bv = (const float*)d_in[6];
  const float* Wo = (const float*)d_in[7];
  const float* bo = (const float*)d_in[8];

  // Workspace layout; xb region reused for attention output. Total 40 MB.
  char* ws = (char*)d_ws;
  const size_t MB = 1u << 20;
  __bf16* xb   = (__bf16*)(ws + 0 * MB);   // 8 MB (4096x1024)
  __bf16* Wqt  = (__bf16*)(ws + 8 * MB);   // 2 MB
  __bf16* Wkt  = (__bf16*)(ws + 10 * MB);  // 2 MB
  __bf16* Wvt  = (__bf16*)(ws + 12 * MB);  // 2 MB
  __bf16* Wot  = (__bf16*)(ws + 14 * MB);  // 2 MB
  __bf16* Qb   = (__bf16*)(ws + 16 * MB);  // 8 MB
  __bf16* Kb   = (__bf16*)(ws + 24 * MB);  // 8 MB
  __bf16* Vtb  = (__bf16*)(ws + 32 * MB);  // 8 MB (b,h,dh,s)
  __bf16* attn = xb;                       // reuse

  mha_cvt_bf16<<<(MTOT * DMODEL) / (8 * 256), 256, 0, stream>>>(x, xb);

  dim3 tb(32, 8), tg(DMODEL / 32, DMODEL / 32);
  mha_transpose_cvt<<<tg, tb, 0, stream>>>(Wq, Wqt);
  mha_transpose_cvt<<<tg, tb, 0, stream>>>(Wk, Wkt);
  mha_transpose_cvt<<<tg, tb, 0, stream>>>(Wv, Wvt);
  mha_transpose_cvt<<<tg, tb, 0, stream>>>(Wo, Wot);

  dim3 gg(DMODEL / 128, MTOT / 128);       // (8, 32)
  mha_gemm_bf16<OUT_BF16><<<gg, 256, 0, stream>>>(xb, Wqt, bq, (void*)Qb);
  mha_gemm_bf16<OUT_BF16><<<gg, 256, 0, stream>>>(xb, Wkt, bk, (void*)Kb);
  mha_gemm_bf16<OUT_VT  ><<<gg, 256, 0, stream>>>(xb, Wvt, bv, (void*)Vtb);

  dim3 ag(SEQ / 128, NBATCH * NHEADS);     // (16, 32)
  mha_attention<<<ag, 256, 0, stream>>>(Qb, Kb, Vtb, attn);

  mha_gemm_bf16<OUT_F32><<<gg, 256, 0, stream>>>(attn, Wot, bo, d_out);
}
